// SimpleAttention_24404004176373
// MI455X (gfx1250) — matrix operand order repaired and checked by reading, not compiled, not run on hardware
//
#include <hip/hip_runtime.h>
#include <hip/hip_bf16.h>

// ---------------------------------------------------------------------------
// SimpleAttention on MI455X (gfx1250), wave32 + WMMA f32_16x16x32_f16.
//
// Shapes: B=4, C=128, H=W=64, N=4096.  Dominated by the two N x N x C
// attention GEMMs (34.4 GFLOP).  Flash-style fusion: never materialize the
// 256MB attention matrix; instead
//   K2: per 16-row m-tile, two WMMA sweeps over n recomputing S=Q'K to get
//       rowmax and 1/rowsum (exp2 domain; scale*log2e folded into q).
//   K4: per 16-col n-tile, sweep m in slabs of 32: recompute two S tiles,
//       apply exp2(s-m)*1/Z, transpose the 16x32 P slab through LDS into
//       B-fragment layout, one WMMA accumulate out += V * P per slab.
//       The q slab (shared by all 8 waves) is double-buffered into LDS with
//       GLOBAL_LOAD_ASYNC_TO_LDS_B128 (ASYNCcnt) when available.
// Workspace ~25 MB.  No atomics -> bitwise deterministic across replays.
// ---------------------------------------------------------------------------

typedef _Float16 f16;
typedef _Float16 v16h __attribute__((ext_vector_type(16)));
typedef _Float16 v8h  __attribute__((ext_vector_type(8)));
typedef _Float16 v2h  __attribute__((ext_vector_type(2)));
typedef float    v8f  __attribute__((ext_vector_type(8)));
typedef int      v4i  __attribute__((ext_vector_type(4)));

constexpr int kB  = 4;
constexpr int kC  = 128;    // channels == head dim
constexpr int kN  = 4096;   // H*W
constexpr int kNT = 256;    // 16-wide tiles along N
constexpr float kScale    = 0.08838834764831845f;  // C^-0.5
constexpr float kLog2e    = 1.4426950408889634f;
constexpr float kScaleL2e = kScale * kLog2e;       // folded into q at projection

#if defined(__has_builtin)
#if __has_builtin(__builtin_amdgcn_exp2f)
#define EXP2F(x) __builtin_amdgcn_exp2f(x)
#endif
#if __has_builtin(__builtin_amdgcn_global_load_async_to_lds_b128) && \
    __has_builtin(__builtin_amdgcn_s_wait_asynccnt)
#define HAVE_ASYNC_LDS 1
#endif
#endif
#ifndef EXP2F
#define EXP2F(x) exp2f(x)
#endif

// D = A*B + C, 16x16x32 f16 -> f32
__device__ __forceinline__ v8f wmma_f16(v16h a, v16h b, v8f c) {
    return __builtin_amdgcn_wmma_f32_16x16x32_f16(false, a, false, b,
                                                  (short)0, c, false, false);
}

// A fragment (16x32, row-major source, rows = lane&15). A lane holds 16
// contiguous K = {2*sel..2*sel+15}, sel = 0 (lo half-wave) or 8 (hi
// half-wave), matching the B fragments. Two contiguous 16B loads (global or LDS).
__device__ __forceinline__ v16h load_afrag(const f16* rowbase, int koff, int selA) {
    const v8h c0 = *reinterpret_cast<const v8h*>(rowbase + koff + 2 * selA);
    const v8h c1 = *reinterpret_cast<const v8h*>(rowbase + koff + 2 * selA + 8);
    v16h r;
#pragma unroll
    for (int i = 0; i < 8; ++i) { r[i] = c0[i]; r[i + 8] = c1[i]; }
    return r;
}

// B fragment (32x16): lane = column, 16 contiguous K halves (sel 0 or 16
// already applied by the caller). Two contiguous 16B loads.
__device__ __forceinline__ v16h load_bfrag(const f16* p) {
    const v8h c0 = *reinterpret_cast<const v8h*>(p);
    const v8h c1 = *reinterpret_cast<const v8h*>(p + 8);
    v16h r;
#pragma unroll
    for (int i = 0; i < 8; ++i) { r[i] = c0[i]; r[i + 8] = c1[i]; }
    return r;
}

// 16B global -> LDS copy, async (ASYNCcnt path) when available.  The builtin
// takes typed int4 pointers: (v4i AS1*, v4i AS3*, imm offset, imm cpol).
__device__ __forceinline__ void copy16_g2l(const f16* g, f16* l) {
#ifdef HAVE_ASYNC_LDS
    typedef __attribute__((address_space(1))) v4i* gp_t;
    typedef __attribute__((address_space(3))) v4i* lp_t;
    __builtin_amdgcn_global_load_async_to_lds_b128((gp_t)g, (lp_t)l, 0, 0);
#else
    *reinterpret_cast<v8h*>(l) = *reinterpret_cast<const v8h*>(g);
#endif
}
__device__ __forceinline__ void wait_async_lds() {
#ifdef HAVE_ASYNC_LDS
    __builtin_amdgcn_s_wait_asynccnt(0);
#endif
}

// --------------------------- K0: weights -> f16 ----------------------------
__global__ void wcvt_kernel(const float* __restrict__ wq, const float* __restrict__ wk,
                            const float* __restrict__ wv, const float* __restrict__ wp,
                            f16* __restrict__ wqH, f16* __restrict__ wkH,
                            f16* __restrict__ wvH, f16* __restrict__ wpH) {
    int i = blockIdx.x * blockDim.x + threadIdx.x;
    if (i < kC * kC) {
        wqH[i] = (f16)wq[i]; wkH[i] = (f16)wk[i];
        wvH[i] = (f16)wv[i]; wpH[i] = (f16)wp[i];
    }
}

// ------------------- K1: q/k/v projections (WMMA) --------------------------
// qh[b][n][c] = (Wq x + bq) * scale * log2e   (f16, transposed for A/B frags)
// kh[b][n][c] = (Wk x + bk)
// vh[b][c][n] = (Wv x + bv)                   (natural layout for A frags)
__global__ __launch_bounds__(256) void proj_qkv_kernel(
    const float* __restrict__ x,
    const float* __restrict__ bq, const float* __restrict__ bk, const float* __restrict__ bv,
    const f16* __restrict__ wqH, const f16* __restrict__ wkH, const f16* __restrict__ wvH,
    f16* __restrict__ qh, f16* __restrict__ kh, f16* __restrict__ vh) {
    const int b = blockIdx.y;
    const int n0 = blockIdx.x * 16;
    const int tid = threadIdx.x;
    const int wave = tid >> 5, lane = tid & 31;
    const int nl = lane & 15;
    const int selA = (lane < 16) ? 0 : 8;
    const int selB = (lane < 16) ? 0 : 16;
    const int rowsel = (lane < 16) ? 0 : 8;

    __shared__ __align__(16) f16 xT[16][kC + 8];   // transposed f16 x tile [n][c]
    __shared__ __align__(16) f16 Tv[8][16][24];    // per-wave v-tile transpose

    for (int r = tid; r < 16 * kC; r += 256) {
        int c = r >> 4, j = r & 15;
        xT[j][c] = (f16)x[((size_t)b * kC + c) * kN + n0 + j];
    }
    __syncthreads();

    const int o0 = wave * 16;                      // wave -> output-channel tile
    v8f cq = {}, ck = {}, cv = {};
#pragma unroll
    for (int kk = 0; kk < 4; ++kk) {
        const int cb = kk * 32;
        v16h a   = load_afrag(&xT[nl][0], cb, selA);
        v16h bqf = load_bfrag(wqH + (o0 + nl) * kC + cb + selB);
        v16h bkf = load_bfrag(wkH + (o0 + nl) * kC + cb + selB);
        v16h bvf = load_bfrag(wvH + (o0 + nl) * kC + cb + selB);
        cq = wmma_f16(a, bqf, cq);
        ck = wmma_f16(a, bkf, ck);
        cv = wmma_f16(a, bvf, cv);
    }
    const float biasq = bq[o0 + nl], biask = bk[o0 + nl], biasv = bv[o0 + nl];

#pragma unroll
    for (int r = 0; r < 8; ++r) {                  // D: rows m = r + rowsel, col = nl
        int m = r + rowsel;
        size_t idx = ((size_t)b * kN + n0 + m) * kC + o0 + nl;
        qh[idx] = (f16)((cq[r] + biasq) * kScaleL2e);
        kh[idx] = (f16)(ck[r] + biask);
    }
    // v: transpose through wave-private LDS so vh[b][c][n] stores are coalesced
#pragma unroll
    for (int r = 0; r < 8; r += 2) {
        v2h p; p[0] = (f16)(cv[r] + biasv); p[1] = (f16)(cv[r + 1] + biasv);
        *reinterpret_cast<v2h*>(&Tv[wave][nl][rowsel + r]) = p;   // T[o_loc][m_loc]
    }
    __builtin_amdgcn_wave_barrier();
    {
        int j = lane & 15, hh = (lane >> 4) * 8;
        v8h row = *reinterpret_cast<const v8h*>(&Tv[wave][j][hh]);
        *reinterpret_cast<v8h*>(&vh[((size_t)b * kC + o0 + j) * kN + n0 + hh]) = row;
    }
}

// ---------------- K2: softmax stats (rowmax + 1/rowsum) --------------------
// Per (b, 16-row m-tile): two WMMA sweeps over all n; exp2 domain.
__global__ __launch_bounds__(256) void softmax_stats_kernel(
    const f16* __restrict__ qh, const f16* __restrict__ kh,
    float* __restrict__ rowM, float* __restrict__ rowR) {
    const int b = blockIdx.y;
    const int m0 = blockIdx.x * 16;
    const int tid = threadIdx.x, wave = tid >> 5, lane = tid & 31;
    const int nl = lane & 15;
    const int selA = (lane < 16) ? 0 : 8;
    const int selB = (lane < 16) ? 0 : 16;
    const int rowsel = (lane < 16) ? 0 : 8;

    __shared__ float wred[8][16];
    __shared__ float rowstat[16];

    v16h qa[4];
    const f16* qrow = qh + ((size_t)b * kN + m0 + nl) * kC;
#pragma unroll
    for (int kk = 0; kk < 4; ++kk) qa[kk] = load_afrag(qrow, kk * 32, selA);

    // ---- pass 1: row max ----
    float vmax[8];
#pragma unroll
    for (int r = 0; r < 8; ++r) vmax[r] = -3.0e38f;
    for (int t = wave; t < kNT; t += 8) {
        const f16* krow = kh + ((size_t)b * kN + t * 16 + nl) * kC;
        v8f s = {};
#pragma unroll
        for (int kk = 0; kk < 4; ++kk)
            s = wmma_f16(qa[kk], load_bfrag(krow + kk * 32 + selB), s);
#pragma unroll
        for (int r = 0; r < 8; ++r) vmax[r] = fmaxf(vmax[r], s[r]);
    }
#pragma unroll
    for (int r = 0; r < 8; ++r)
        for (int msk = 1; msk < 16; msk <<= 1)
            vmax[r] = fmaxf(vmax[r], __shfl_xor(vmax[r], msk, 32));
    if (lane == 0)  { for (int r = 0; r < 8; ++r) wred[wave][r]     = vmax[r]; }
    if (lane == 16) { for (int r = 0; r < 8; ++r) wred[wave][8 + r] = vmax[r]; }
    __syncthreads();
    if (tid < 16) {
        float m = wred[0][tid];
        for (int w = 1; w < 8; ++w) m = fmaxf(m, wred[w][tid]);
        rowstat[tid] = m;
    }
    __syncthreads();

    // ---- pass 2: row sum of exp2(s - max) ----
    float rmv[8], vsum[8];
#pragma unroll
    for (int r = 0; r < 8; ++r) { rmv[r] = rowstat[r + rowsel]; vsum[r] = 0.0f; }
    for (int t = wave; t < kNT; t += 8) {
        const f16* krow = kh + ((size_t)b * kN + t * 16 + nl) * kC;
        v8f s = {};
#pragma unroll
        for (int kk = 0; kk < 4; ++kk)
            s = wmma_f16(qa[kk], load_bfrag(krow + kk * 32 + selB), s);
#pragma unroll
        for (int r = 0; r < 8; ++r) vsum[r] += EXP2F(s[r] - rmv[r]);
    }
#pragma unroll
    for (int r = 0; r < 8; ++r)
        for (int msk = 1; msk < 16; msk <<= 1)
            vsum[r] += __shfl_xor(vsum[r], msk, 32);
    if (lane == 0)  { for (int r = 0; r < 8; ++r) wred[wave][r]     = vsum[r]; }
    if (lane == 16) { for (int r = 0; r < 8; ++r) wred[wave][8 + r] = vsum[r]; }
    __syncthreads();
    if (tid < 16) {
        float z = 0.0f;
        for (int w = 0; w < 8; ++w) z += wred[w][tid];
        rowM[(size_t)b * kN + m0 + tid] = rowstat[tid];
        rowR[(size_t)b * kN + m0 + tid] = 1.0f / z;
    }
}

// ----------- K4: fused  hT = (V * softmax(Q'K))^T  (flash-style) ----------
// Block = (b, 16-col n-tile); wave w owns c-tile w.  Sweep m in slabs of 32.
// The 32x128 q slab is identical for all 8 waves -> double-buffered into LDS
// with async global->LDS loads; A-frags then come from conflict-free ds reads.
__global__ __launch_bounds__(256) void attn_av_kernel(
    const f16* __restrict__ qh, const f16* __restrict__ kh, const f16* __restrict__ vh,
    const float* __restrict__ rowM, const float* __restrict__ rowR,
    f16* __restrict__ hT) {
    const int b = blockIdx.y;
    const int n0 = blockIdx.x * 16;
    const int tid = threadIdx.x, wave = tid >> 5, lane = tid & 31;
    const int nl = lane & 15;
    const int selA = (lane < 16) ? 0 : 8;
    const int selB = (lane < 16) ? 0 : 16;
    const int rowsel = (lane < 16) ? 0 : 8;

    __shared__ float sM[kN];                       // rowmax (exp2 domain)
    __shared__ float sR[kN];                       // 1/rowsum
    __shared__ __align__(16) f16 Tp[8][16][48];    // per-wave P slab [n_loc][m 0..31]
    __shared__ __align__(16) f16 Tq[2][32][kC + 8];// double-buffered q slab [m_loc][c]

    for (int i = tid; i < kN; i += 256) {
        sM[i] = rowM[(size_t)b * kN + i];
        sR[i] = rowR[(size_t)b * kN + i];
    }

    const f16* qbase = qh + (size_t)b * kN * kC;
    const int srow = tid >> 4;                     // 0..15  (16B chunk row within half-slab)
    const int sseg = tid & 15;                     // 0..15  (16B chunk within a 256B row)
    // stage rows [m0, m0+32) of q into Tq[buf]
    auto stage_q = [&](int buf, int m0) {
#pragma unroll
        for (int r = 0; r < 2; ++r) {
            int row = r * 16 + srow;
            copy16_g2l(qbase + (size_t)(m0 + row) * kC + sseg * 8,
                       &Tq[buf][row][sseg * 8]);
        }
    };

    stage_q(0, 0);                                 // prologue: slab 0 in flight
    wait_async_lds();
    __syncthreads();                               // sM/sR + slab 0 visible

    const int c0 = wave * 16;
    v16h kb[4];                                    // K frags for this n-tile: resident
    const f16* krow = kh + ((size_t)b * kN + n0 + nl) * kC;
#pragma unroll
    for (int kk = 0; kk < 4; ++kk) kb[kk] = load_bfrag(krow + kk * 32 + selB);

    const f16* vrow = vh + ((size_t)b * kC + c0 + nl) * kN;
    v8f acc = {};

    for (int m0 = 0; m0 < kN; m0 += 32) {
        const int cur = (m0 >> 5) & 1;
        if (m0 + 32 < kN) stage_q(cur ^ 1, m0 + 32);   // overlap DMA with compute
        __builtin_prefetch(vrow + m0 + 64, 0, 1);      // global_prefetch_b8

        v8f s0 = {}, s1 = {};
#pragma unroll
        for (int kk = 0; kk < 4; ++kk) {
            s0 = wmma_f16(load_afrag(&Tq[cur][nl][0],      kk * 32, selA), kb[kk], s0);
            s1 = wmma_f16(load_afrag(&Tq[cur][16 + nl][0], kk * 32, selA), kb[kk], s1);
        }
        // normalized P, transposed into B-frag layout: Tp[n_loc][m_loc]
#pragma unroll
        for (int r = 0; r < 8; r += 2) {
            v2h p, p2;
            int mm;
            mm = m0 + rowsel + r;          p[0]  = (f16)(EXP2F(s0[r]     - sM[mm]) * sR[mm]);
            mm = m0 + rowsel + r + 1;      p[1]  = (f16)(EXP2F(s0[r + 1] - sM[mm]) * sR[mm]);
            mm = m0 + 16 + rowsel + r;     p2[0] = (f16)(EXP2F(s1[r]     - sM[mm]) * sR[mm]);
            mm = m0 + 16 + rowsel + r + 1; p2[1] = (f16)(EXP2F(s1[r + 1] - sM[mm]) * sR[mm]);
            *reinterpret_cast<v2h*>(&Tp[wave][nl][rowsel + r])      = p;
            *reinterpret_cast<v2h*>(&Tp[wave][nl][16 + rowsel + r]) = p2;
        }
        __builtin_amdgcn_wave_barrier();
        v16h pb;
        {
            const v8h* pp = reinterpret_cast<const v8h*>(&Tp[wave][nl][selB]);
            v8h h0 = pp[0], h1 = pp[1];
#pragma unroll
            for (int i = 0; i < 8; ++i) { pb[i] = h0[i]; pb[8 + i] = h1[i]; }
        }
        v16h va = load_afrag(vrow + m0, 0, selA);  // V rows c, K = m slab
        acc = wmma_f16(va, pb, acc);

        wait_async_lds();                          // next slab landed during compute
        __syncthreads();                           // all waves done with cur, next visible
    }
    // transpose 16x16 f32 acc -> hT[b][n][c] f16 via LDS
#pragma unroll
    for (int r = 0; r < 8; r += 2) {
        v2h p; p[0] = (f16)acc[r]; p[1] = (f16)acc[r + 1];
        *reinterpret_cast<v2h*>(&Tp[wave][nl][rowsel + r]) = p;
    }
    __builtin_amdgcn_wave_barrier();
    {
        int j = lane & 15, hh = (lane >> 4) * 8;
        v8h row = *reinterpret_cast<const v8h*>(&Tp[wave][j][hh]);
        *reinterpret_cast<v8h*>(&hT[((size_t)b * kN + n0 + j) * kC + c0 + hh]) = row;
    }
}

// ----------------- K5: output 1x1 conv (WMMA) + bias -----------------------
// o2[b][o][n] = sum_c wp[o][c] * h[c][n] + bp[o]   (f32)
__global__ __launch_bounds__(256) void proj_out_kernel(
    const f16* __restrict__ wpH, const float* __restrict__ bp,
    const f16* __restrict__ hT, float* __restrict__ o2) {
    const int b = blockIdx.y;
    const int n0 = blockIdx.x * 16;
    const int tid = threadIdx.x, wave = tid >> 5, lane = tid & 31;
    const int nl = lane & 15;
    const int selA = (lane < 16) ? 0 : 8;
    const int selB = (lane < 16) ? 0 : 16;
    const int rowsel = (lane < 16) ? 0 : 8;
    const int o0 = wave * 16;

    v16h aw[4];
    const f16* wrow = wpH + (o0 + nl) * kC;
#pragma unroll
    for (int kk = 0; kk < 4; ++kk) aw[kk] = load_afrag(wrow, kk * 32, selA);

    const f16* hrow = hT + ((size_t)b * kN + n0 + nl) * kC;
    v8f acc = {};
#pragma unroll
    for (int kk = 0; kk < 4; ++kk)
        acc = wmma_f16(aw[kk], load_bfrag(hrow + kk * 32 + selB), acc);

#pragma unroll
    for (int r = 0; r < 8; ++r) {
        int o = o0 + rowsel + r;
        o2[((size_t)b * kC + o) * kN + n0 + nl] = acc[r] + bp[o];
    }
}

// --------------- K6: InstanceNorm (biased var) + residual ------------------
__global__ __launch_bounds__(256) void inorm_res_kernel(
    const float* __restrict__ o2, const float* __restrict__ x, float* __restrict__ out) {
    const int row = blockIdx.x;                    // b*C + c
    const int tid = threadIdx.x;
    __shared__ float rs[256], rq[256];
    const float* p = o2 + (size_t)row * kN;
    float s = 0.0f, ss = 0.0f;
    for (int i = tid; i < kN; i += 256) { float v = p[i]; s += v; ss += v * v; }
    rs[tid] = s; rq[tid] = ss;
    __syncthreads();
    for (int st = 128; st > 0; st >>= 1) {
        if (tid < st) { rs[tid] += rs[tid + st]; rq[tid] += rq[tid + st]; }
        __syncthreads();
    }
    const float mean = rs[0] * (1.0f / kN);
    const float var  = rq[0] * (1.0f / kN) - mean * mean;
    const float rstd = rsqrtf(var + 1e-5f);
    const float* xr = x + (size_t)row * kN;
    float* orow = out + (size_t)row * kN;
    for (int i = tid; i < kN; i += 256) orow[i] = (p[i] - mean) * rstd + xr[i];
}

// ---------------------------------------------------------------------------
extern "C" void kernel_launch(void* const* d_in, const int* in_sizes, int n_in,
                              void* d_out, int out_size, void* d_ws, size_t ws_size,
                              hipStream_t stream) {
    (void)in_sizes; (void)n_in; (void)out_size; (void)ws_size;
    const float* x  = (const float*)d_in[0];
    const float* wq = (const float*)d_in[1];
    const float* bq = (const float*)d_in[2];
    const float* wk = (const float*)d_in[3];
    const float* bk = (const float*)d_in[4];
    const float* wv = (const float*)d_in[5];
    const float* bv = (const float*)d_in[6];
    const float* wp = (const float*)d_in[7];
    const float* bp = (const float*)d_in[8];
    float* out = (float*)d_out;

    char* ws = (char*)d_ws;
    constexpr size_t SZ_W   = (size_t)kC * kC * sizeof(f16);          // 32 KB
    constexpr size_t SZ_QKV = (size_t)kB * kN * kC * sizeof(f16);     // 4 MB
    constexpr size_t SZ_ROW = (size_t)kB * kN * sizeof(float);        // 64 KB
    f16*   wqH  = (f16*)(ws);
    f16*   wkH  = (f16*)(ws + SZ_W);
    f16*   wvH  = (f16*)(ws + 2 * SZ_W);
    f16*   wpH  = (f16*)(ws + 3 * SZ_W);
    f16*   qh   = (f16*)(ws + 4 * SZ_W);
    f16*   kh   = (f16*)(ws + 4 * SZ_W + SZ_QKV);
    f16*   vh   = (f16*)(ws + 4 * SZ_W + 2 * SZ_QKV);
    f16*   hT   = (f16*)(ws + 4 * SZ_W + 3 * SZ_QKV);
    float* rowM = (float*)(ws + 4 * SZ_W + 4 * SZ_QKV);
    float* rowR = (float*)(ws + 4 * SZ_W + 4 * SZ_QKV + SZ_ROW);
    float* o2   = (float*)(ws + 4 * SZ_W + 4 * SZ_QKV + 2 * SZ_ROW);  // +8 MB

    wcvt_kernel<<<dim3((kC * kC + 255) / 256), 256, 0, stream>>>(
        wq, wk, wv, wp, wqH, wkH, wvH, wpH);
    proj_qkv_kernel<<<dim3(kNT, kB), 256, 0, stream>>>(
        x, bq, bk, bv, wqH, wkH, wvH, qh, kh, vh);
    softmax_stats_kernel<<<dim3(kNT, kB), 256, 0, stream>>>(qh, kh, rowM, rowR);
    attn_av_kernel<<<dim3(kNT, kB), 256, 0, stream>>>(qh, kh, vh, rowM, rowR, hT);
    proj_out_kernel<<<dim3(kNT, kB), 256, 0, stream>>>(wpH, bp, hT, o2);
    inorm_res_kernel<<<dim3(kB * kC), 256, 0, stream>>>(o2, x, out);
}
